// MoeLayer_33148557590839
// MI455X (gfx1250) — matrix operand order repaired and checked by reading, not compiled, not run on hardware
//
#include <hip/hip_runtime.h>
#include <hip/hip_bf16.h>

#define B_  4
#define S_  2048
#define D_  1024
#define H_  1024
#define E_  8
#define NT  (B_ * S_)   // 8192 tokens

typedef __attribute__((ext_vector_type(16))) __bf16 v16bf;
typedef __attribute__((ext_vector_type(8)))  __bf16 v8bf;
typedef __attribute__((ext_vector_type(8)))  float  v8f;

__device__ __forceinline__ unsigned short f2bfu(float f) {
    union { float f; unsigned u; } a; a.f = f;
    return (unsigned short)((a.u + 0x7FFFu + ((a.u >> 16) & 1u)) >> 16);   // RNE
}

// ---------------------------------------------------------------------------
// One-shot: x (f32, [NT][D]) -> bf16 row-major copy in workspace.
// ---------------------------------------------------------------------------
__global__ __launch_bounds__(256)
void conv_x(const float* __restrict__ x, unsigned short* __restrict__ xb) {
    const int idx = blockIdx.x * 256 + threadIdx.x;          // one float4 each
    const float4 v = ((const float4*)x)[idx];
    uint2 o;
    o.x = (unsigned)f2bfu(v.x) | ((unsigned)f2bfu(v.y) << 16);
    o.y = (unsigned)f2bfu(v.z) | ((unsigned)f2bfu(v.w) << 16);
    ((uint2*)xb)[idx] = o;
}

// ---------------------------------------------------------------------------
// One-shot: We (f32, [E][D][H]) -> bf16 transposed [E][H][D] (K-major),
// 32x32 tiles through LDS so both global sides stay coalesced.
// ---------------------------------------------------------------------------
__global__ __launch_bounds__(256)
void conv_we(const float* __restrict__ We, unsigned short* __restrict__ Wt) {
    __shared__ unsigned short t[32][33];
    const int e  = blockIdx.z;
    const int d0 = blockIdx.x * 32;
    const int h0 = blockIdx.y * 32;
    const int tid = threadIdx.x;

    { // read: rows of We along h (coalesced)
        const int d  = tid >> 3;
        const int hq = (tid & 7) * 4;
        const float4 v = *(const float4*)(We + ((size_t)e * D_ + d0 + d) * H_ + h0 + hq);
        t[d][hq + 0] = f2bfu(v.x);
        t[d][hq + 1] = f2bfu(v.y);
        t[d][hq + 2] = f2bfu(v.z);
        t[d][hq + 3] = f2bfu(v.w);
    }
    __syncthreads();
    { // write: rows of Wt along d (coalesced 8B stores)
        const int h  = tid >> 3;
        const int dq = (tid & 7) * 4;
        uint2 o;
        o.x = (unsigned)t[dq + 0][h] | ((unsigned)t[dq + 1][h] << 16);
        o.y = (unsigned)t[dq + 2][h] | ((unsigned)t[dq + 3][h] << 16);
        *(uint2*)(Wt + ((size_t)e * H_ + h0 + h) * D_ + d0 + dq) = o;
    }
}

// ---------------------------------------------------------------------------
// Gate: one wave32 per token. logits = x@Wg + bg; top-2 + softmax; append
// (token, weight) to per-(slot, expert) routing lists via atomics.
// ---------------------------------------------------------------------------
__global__ __launch_bounds__(256)
void moe_gate(const float* __restrict__ x, const float* __restrict__ Wg,
              const float* __restrict__ bg,
              int* __restrict__ counts, int* __restrict__ toks,
              float* __restrict__ wgts) {
    const int tok  = (blockIdx.x * blockDim.x + threadIdx.x) >> 5;
    const int lane = threadIdx.x & 31;
    if (tok >= NT) return;

    const float* xr = x + (size_t)tok * D_;
    float acc[E_];
#pragma unroll
    for (int e = 0; e < E_; ++e) acc[e] = 0.f;

    for (int d = lane; d < D_; d += 32) {
        const float xv = xr[d];
        const float* wr = Wg + d * E_;
#pragma unroll
        for (int e = 0; e < E_; ++e) acc[e] += xv * wr[e];
    }
#pragma unroll
    for (int off = 16; off > 0; off >>= 1) {
#pragma unroll
        for (int e = 0; e < E_; ++e) acc[e] += __shfl_down(acc[e], off, 32);
    }

    if (lane == 0) {
        float v[E_];
#pragma unroll
        for (int e = 0; e < E_; ++e) v[e] = acc[e] + bg[e];
        int e0 = 0; float v0 = v[0];
#pragma unroll
        for (int e = 1; e < E_; ++e) if (v[e] > v0) { v0 = v[e]; e0 = e; }
        int e1 = (e0 == 0) ? 1 : 0; float v1 = v[e1];
#pragma unroll
        for (int e = 0; e < E_; ++e)
            if (e != e0 && e != e1 && v[e] > v1) { v1 = v[e]; e1 = e; }
        const float z  = __expf(v1 - v0);
        const float w0 = 1.f / (1.f + z);
        const float w1 = z   / (1.f + z);

        int p0 = atomicAdd(&counts[e0], 1);
        toks[(size_t)e0 * NT + p0] = tok;
        wgts[(size_t)e0 * NT + p0] = w0;
        int p1 = atomicAdd(&counts[E_ + e1], 1);
        toks[(size_t)(E_ + e1) * NT + p1] = tok;
        wgts[(size_t)(E_ + e1) * NT + p1] = w1;
    }
}

// ---------------------------------------------------------------------------
// Expert GEMM over gathered token lists (bf16 WMMA, f32 accumulate).
// Block: 256 threads = 8 waves; tile = 16 tokens x 128 H-cols.
// A strip (16 tokens x full D, bf16, 32 KB) preloaded to LDS once -> the
// K-loop has no barriers: per 64-K step each wave does 4 ds_load_b128 (A),
// 4 global_load_b128 (B, K-major We_t, pure L2-resident stream), 2 WMMAs.
// slot 0 stores, slot 1 accumulates (stream-serialized, no atomics).
// ---------------------------------------------------------------------------
__global__ __launch_bounds__(256)
void moe_expert_gemm(const unsigned short* __restrict__ xb,
                     const unsigned short* __restrict__ Wt,
                     const float* __restrict__ be,
                     const int* __restrict__ counts, const int* __restrict__ toks,
                     const float* __restrict__ wgts,
                     float* __restrict__ out, int slot) {
    const int e     = blockIdx.z;
    const int list  = slot * E_ + e;
    const int count = counts[list];
    const int tile0 = blockIdx.x * 16;
    if (tile0 >= count) return;          // uniform exit: EXEC stays full for WMMA

    __shared__ unsigned short lA[16 * D_];   // 32 KB: 16 gathered token rows, bf16
    __shared__ int    sTok[16];
    __shared__ float  sW[16];

    const int tid = threadIdx.x;
    if (tid < 16) {
        const int gr    = tile0 + tid;
        const int valid = gr < count;
        sTok[tid] = toks[(size_t)list * NT + (valid ? gr : tile0)];
        sW[tid]   = valid ? wgts[(size_t)list * NT + gr] : 0.f;
    }
    __syncthreads();

    { // preload A strip: each thread copies 64 bf16 (8 x 16B), coalesced
        const int row = tid >> 4;
        const int seg = (tid & 15) * 64;
        const unsigned short* src = xb + (size_t)sTok[row] * D_ + seg;
        unsigned short*       dst = lA + row * D_ + seg;
#pragma unroll
        for (int i = 0; i < 8; ++i)
            *(uint4*)(dst + i * 8) = *(const uint4*)(src + i * 8);
    }
    __syncthreads();

    const int hbase = blockIdx.y * 128;
    const int lane  = tid & 31;
    const int wid   = tid >> 5;
    const int colT  = lane & 15;
    const int half  = lane >> 4;
    const int kbA   = half * 16;         // A frag: lanes<16 -> K+0-15; else K+16-31
    const int kbB   = half * 16;         // B frag: lanes<16 -> K+0-15; else K+16-31
    const int arow  = lane & 15;

    const unsigned short* arowp = &lA[arow * D_];
    const unsigned short* bp    = Wt + ((size_t)e * H_ + hbase + wid * 16 + colT) * (size_t)D_;

    v8f c = {};

#pragma unroll 4
    for (int d0 = 0; d0 < D_; d0 += 64) {
        __builtin_prefetch(bp + d0 + 256, 0, 0);
        // B fragments straight from global (K-major, 32B aligned)
        const v16bf b0 = *(const v16bf*)(bp + d0 + kbB);
        const v16bf b1 = *(const v16bf*)(bp + d0 + 32 + kbB);
        // A fragments from LDS (broadcast across the 8 waves)
        const v8bf a0lo = *(const v8bf*)(arowp + d0 + kbA);
        const v8bf a0hi = *(const v8bf*)(arowp + d0 + kbA + 8);
        const v8bf a1lo = *(const v8bf*)(arowp + d0 + 32 + kbA);
        const v8bf a1hi = *(const v8bf*)(arowp + d0 + 32 + kbA + 8);
        v16bf a0, a1;
#pragma unroll
        for (int i = 0; i < 8; ++i) {
            a0[i] = a0lo[i]; a0[i + 8] = a0hi[i];
            a1[i] = a1lo[i]; a1[i + 8] = a1hi[i];
        }
        c = __builtin_amdgcn_wmma_f32_16x16x32_bf16(false, a0, false, b0,
                                                    (short)0, c, false, false);
        c = __builtin_amdgcn_wmma_f32_16x16x32_bf16(false, a1, false, b1,
                                                    (short)0, c, false, false);
    }

    // epilogue: weight * (acc + bias), scatter to gathered token rows
    const int   col = hbase + wid * 16 + colT;
    const float bev = be[e * H_ + col];
#pragma unroll
    for (int r = 0; r < 8; ++r) {
        const int M  = r + half * 8;     // C/D layout: lanes 0-15 -> M 0-7; 16-31 -> M 8-15
        const int gr = tile0 + M;
        if (gr < count) {
            const float val = (c[r] + bev) * sW[M];
            float* op = out + (size_t)sTok[M] * H_ + col;
            if (slot == 0) *op = val;
            else           *op += val;
        }
    }
}

extern "C" void kernel_launch(void* const* d_in, const int* in_sizes, int n_in,
                              void* d_out, int out_size, void* d_ws, size_t ws_size,
                              hipStream_t stream) {
    (void)in_sizes; (void)n_in; (void)out_size; (void)ws_size;
    const float* x  = (const float*)d_in[0];
    const float* Wg = (const float*)d_in[1];
    const float* bg = (const float*)d_in[2];
    const float* We = (const float*)d_in[3];
    const float* be = (const float*)d_in[4];
    // d_in[5] = k (== 2, baked in)

    // workspace layout (~34.6 MB)
    char* ws = (char*)d_ws;
    int*            counts = (int*)ws;                                  // 2*E ints
    size_t off = 256;
    int*            toks   = (int*)(ws + off);    off += sizeof(int)   * 2 * E_ * NT;
    float*          wgts   = (float*)(ws + off);  off += sizeof(float) * 2 * E_ * NT;
    unsigned short* xb     = (unsigned short*)(ws + off); off += (size_t)NT * D_ * 2;
    unsigned short* Wt     = (unsigned short*)(ws + off);
    float* out = (float*)d_out;

    hipMemsetAsync(counts, 0, sizeof(int) * 2 * E_, stream);

    conv_x<<<(NT * D_ / 4) / 256, 256, 0, stream>>>(x, xb);
    conv_we<<<dim3(D_ / 32, H_ / 32, E_), 256, 0, stream>>>(We, Wt);
    moe_gate<<<NT / 8, 256, 0, stream>>>(x, Wg, bg, counts, toks, wgts);

    dim3 grid(NT / 16, H_ / 128, E_);
    moe_expert_gemm<<<grid, 256, 0, stream>>>(xb, Wt, be, counts, toks, wgts, (float*)d_out, 0);
    moe_expert_gemm<<<grid, 256, 0, stream>>>(xb, Wt, be, counts, toks, wgts, (float*)d_out, 1);
    (void)out;
}
